// MultiHeadAttentionWithWeights_24395414241619
// MI455X (gfx1250) — matrix operand order repaired and checked by reading, not compiled, not run on hardware
//
#include <hip/hip_runtime.h>

typedef __bf16 bf16;
typedef __attribute__((ext_vector_type(16))) __bf16 v16bf;
typedef __attribute__((ext_vector_type(8)))  float  v8f;

// ---------------------------------------------------------------------------
// WMMA helpers (CDNA5 v_wmma_f32_16x16x32_bf16, wave32)
// ---------------------------------------------------------------------------
__device__ __forceinline__ v8f wmma_bf16(v16bf a, v16bf b, v8f c) {
  return __builtin_amdgcn_wmma_f32_16x16x32_bf16(
      /*neg_a=*/false, a, /*neg_b=*/false, b,
      /*c_mod=*/(short)0, c, /*reuse_a=*/false, /*reuse_b=*/false);
}

// A fragment: 16x32 bf16 tile from LDS (row-major, ld=32).
// Lanes 0-15 hold M; lane half h holds k = 16h + s (contiguous, matches frag_b).
// Lane data = row[khi..khi+15]  -> two ds_load_b128.
__device__ __forceinline__ v16bf frag_a(const bf16* ldsA, int mBase) {
  const int l   = threadIdx.x & 31;
  const int m   = mBase + (l & 15);
  const int khi = (l >> 4) << 4;
  const bf16* row = ldsA + m * 32 + khi;
  v16bf a;
  *reinterpret_cast<uint4*>(&a)       = *reinterpret_cast<const uint4*>(row);
  *(reinterpret_cast<uint4*>(&a) + 1) = *reinterpret_cast<const uint4*>(row + 8);
  return a;
}

// B fragment from N-major LDS tile ldsBT[n][32]: one contiguous 32B read.
__device__ __forceinline__ v16bf frag_b(const bf16* ldsBT, int nBase) {
  const int l  = threadIdx.x & 31;
  const int n  = nBase + (l & 15);
  const int kb = (l >> 4) << 4;
  return *reinterpret_cast<const v16bf*>(ldsBT + n * 32 + kb);
}

// 16 fp32 -> 16 bf16 into LDS (two contiguous 16B stores)
__device__ __forceinline__ void cvt16_store(const float4* __restrict__ src,
                                            bf16* __restrict__ dst) {
  bf16 tmp[16];
#pragma unroll
  for (int j = 0; j < 4; ++j) {
    float4 f = src[j];
    tmp[4 * j]     = (bf16)f.x;
    tmp[4 * j + 1] = (bf16)f.y;
    tmp[4 * j + 2] = (bf16)f.z;
    tmp[4 * j + 3] = (bf16)f.w;
  }
  reinterpret_cast<uint4*>(dst)[0] = reinterpret_cast<uint4*>(tmp)[0];
  reinterpret_cast<uint4*>(dst)[1] = reinterpret_cast<uint4*>(tmp)[1];
}

// Async global->LDS copy of 16 bf16 (32B) per lane, no VGPR staging.
// GLOBAL_LOAD_ASYNC_TO_LDS_B128: VDST = LDS byte address (flat addr low 32b),
// VADDR = 64-bit global address; INST_OFFSET added to both sides (ISA 10.x).
// Tracked by ASYNCcnt.
__device__ __forceinline__ void async_copy16bf(const bf16* __restrict__ src,
                                               bf16* __restrict__ ldsDst) {
  const unsigned ldsOff = (unsigned)(unsigned long long)(uintptr_t)ldsDst;
  asm volatile(
      "global_load_async_to_lds_b128 %0, %1, off\n\t"
      "global_load_async_to_lds_b128 %0, %1, off offset:16"
      :
      : "v"(ldsOff), "v"(src)
      : "memory");
}

__device__ __forceinline__ void wait_async0() {
  asm volatile("s_wait_asynccnt 0x0" ::: "memory");
}

// ---------------------------------------------------------------------------
// Kernel 1: qkv = x @ W_qkv^T + b ; scatter to bf16 Q/K/V [B,H,S,D]
//   M=4096, N=2304, K=768. Block 128x128; wave tile 32x64 (8 wmma/K-step).
// ---------------------------------------------------------------------------
__global__ __launch_bounds__(256) void qkv_gemm(
    const float* __restrict__ x, const float* __restrict__ W,
    const float* __restrict__ bias,
    bf16* __restrict__ Q, bf16* __restrict__ Kt, bf16* __restrict__ V) {
  __shared__ alignas(32) bf16 ldsA[128 * 32];   // M-major
  __shared__ alignas(32) bf16 ldsBT[128 * 32];  // N-major
  const int tid = threadIdx.x, lane = tid & 31, wave = tid >> 5;
  const int wm = wave >> 1, wn = wave & 1;
  const int row0 = blockIdx.y * 128;
  const int col0 = blockIdx.x * 128;
  v8f acc[2][4] = {};

  for (int k0 = 0; k0 < 768; k0 += 32) {
    const int r = tid >> 1, c = (tid & 1) << 4;
    cvt16_store(reinterpret_cast<const float4*>(x + (size_t)(row0 + r) * 768 + k0 + c),
                ldsA + r * 32 + c);
    cvt16_store(reinterpret_cast<const float4*>(W + (size_t)(col0 + r) * 768 + k0 + c),
                ldsBT + r * 32 + c);
    __syncthreads();
    v16bf a0 = frag_a(ldsA, wm * 32);
    v16bf a1 = frag_a(ldsA, wm * 32 + 16);
#pragma unroll
    for (int t = 0; t < 4; ++t) {
      v16bf bfr = frag_b(ldsBT, wn * 64 + t * 16);
      acc[0][t] = wmma_bf16(a0, bfr, acc[0][t]);
      acc[1][t] = wmma_bf16(a1, bfr, acc[1][t]);
    }
    __syncthreads();
  }

  const int nl = lane & 15, mh = (lane >> 4) << 3;
#pragma unroll
  for (int u = 0; u < 2; ++u) {
#pragma unroll
    for (int t = 0; t < 4; ++t) {
      const int f = col0 + wn * 64 + t * 16 + nl;
      const float bv = bias[f];
      const int which = f / 768;
      const int h = (f % 768) >> 6;
      const int d = f & 63;
      bf16* dst = (which == 0) ? Q : (which == 1) ? Kt : V;
#pragma unroll
      for (int r = 0; r < 8; ++r) {
        const int tok = row0 + wm * 32 + u * 16 + mh + r;
        const int b = tok >> 11, s = tok & 2047;
        dst[((size_t)((b * 12 + h) * 2048 + s)) * 64 + d] = (bf16)(acc[u][t][r] + bv);
      }
    }
  }
}

// ---------------------------------------------------------------------------
// Kernel 2: raw scores = (Q @ K^T) / 8, masked, written fp32 to attn region.
//   Per (b,h): M=2048, N=2048, K=64. Block 128x128.
//   Tiles are pure bf16 copies -> async global->LDS DMA (no VGPR staging).
// ---------------------------------------------------------------------------
__global__ __launch_bounds__(256) void scores_gemm(
    const bf16* __restrict__ Q, const bf16* __restrict__ Kt,
    const unsigned char* __restrict__ mask, float* __restrict__ attn) {
  __shared__ alignas(32) bf16 ldsA[128 * 32];
  __shared__ alignas(32) bf16 ldsBT[128 * 32];
  const int tid = threadIdx.x, lane = tid & 31, wave = tid >> 5;
  const int wm = wave >> 1, wn = wave & 1;
  const int bh = blockIdx.z;
  const int row0 = blockIdx.y * 128;
  const int col0 = blockIdx.x * 128;
  const bf16* Qb = Q + (size_t)bh * 2048 * 64;
  const bf16* Kb = Kt + (size_t)bh * 2048 * 64;
  v8f acc[2][4] = {};

  for (int k0 = 0; k0 < 64; k0 += 32) {
    const int r = tid >> 1, c = (tid & 1) << 4;
    async_copy16bf(Qb + (size_t)(row0 + r) * 64 + k0 + c, ldsA + r * 32 + c);
    async_copy16bf(Kb + (size_t)(col0 + r) * 64 + k0 + c, ldsBT + r * 32 + c);
    wait_async0();
    __syncthreads();
    v16bf a0 = frag_a(ldsA, wm * 32);
    v16bf a1 = frag_a(ldsA, wm * 32 + 16);
#pragma unroll
    for (int t = 0; t < 4; ++t) {
      v16bf bfr = frag_b(ldsBT, wn * 64 + t * 16);
      acc[0][t] = wmma_bf16(a0, bfr, acc[0][t]);
      acc[1][t] = wmma_bf16(a1, bfr, acc[1][t]);
    }
    __syncthreads();
  }

  const int b = bh / 12;
  const int nl = lane & 15, mh = (lane >> 4) << 3;
#pragma unroll
  for (int u = 0; u < 2; ++u) {
#pragma unroll
    for (int t = 0; t < 4; ++t) {
      const int kcol = col0 + wn * 64 + t * 16 + nl;
#pragma unroll
      for (int r = 0; r < 8; ++r) {
        const int q = row0 + wm * 32 + u * 16 + mh + r;
        float val = acc[u][t][r] * 0.125f;  // 1/sqrt(64)
        if (mask[((size_t)b * 2048 + q) * 2048 + kcol]) val = -1e9f;
        attn[((size_t)bh * 2048 + q) * 2048 + kcol] = val;
      }
    }
  }
}

// ---------------------------------------------------------------------------
// Kernel 3: in-place row softmax over 2048 elements. One block per row.
// ---------------------------------------------------------------------------
__global__ __launch_bounds__(256) void softmax_rows(float* __restrict__ attn) {
  __shared__ float red[8];
  float* p = attn + (size_t)blockIdx.x * 2048;
  const int tid = threadIdx.x, lane = tid & 31, wave = tid >> 5;
  float4* p4 = reinterpret_cast<float4*>(p);
  float4 a = p4[2 * tid], b = p4[2 * tid + 1];

  float m = fmaxf(fmaxf(fmaxf(a.x, a.y), fmaxf(a.z, a.w)),
                  fmaxf(fmaxf(b.x, b.y), fmaxf(b.z, b.w)));
#pragma unroll
  for (int off = 16; off > 0; off >>= 1) m = fmaxf(m, __shfl_xor(m, off, 32));
  if (lane == 0) red[wave] = m;
  __syncthreads();
  m = red[0];
#pragma unroll
  for (int w = 1; w < 8; ++w) m = fmaxf(m, red[w]);
  __syncthreads();

  a.x = __expf(a.x - m); a.y = __expf(a.y - m); a.z = __expf(a.z - m); a.w = __expf(a.w - m);
  b.x = __expf(b.x - m); b.y = __expf(b.y - m); b.z = __expf(b.z - m); b.w = __expf(b.w - m);
  float s = a.x + a.y + a.z + a.w + b.x + b.y + b.z + b.w;
#pragma unroll
  for (int off = 16; off > 0; off >>= 1) s += __shfl_xor(s, off, 32);
  if (lane == 0) red[wave] = s;
  __syncthreads();
  s = red[0];
#pragma unroll
  for (int w = 1; w < 8; ++w) s += red[w];
  const float inv = 1.0f / s;
  a.x *= inv; a.y *= inv; a.z *= inv; a.w *= inv;
  b.x *= inv; b.y *= inv; b.z *= inv; b.w *= inv;
  p4[2 * tid] = a;
  p4[2 * tid + 1] = b;
}

// ---------------------------------------------------------------------------
// Kernel 4: attn @ V -> bf16 attn-out in [B,S,E] layout.
//   Per (b,h): M=2048, N=64, K=2048. Block 256x64, wave tile 32x64.
// ---------------------------------------------------------------------------
__global__ __launch_bounds__(256) void av_gemm(
    const float* __restrict__ attn, const bf16* __restrict__ V,
    bf16* __restrict__ AO) {
  __shared__ alignas(32) bf16 ldsA[256 * 32];  // M-major
  __shared__ alignas(32) bf16 ldsBT[64 * 32];  // N-major (transposed V)
  const int tid = threadIdx.x, lane = tid & 31, wave = tid >> 5;
  const int bh = blockIdx.y;
  const int row0 = blockIdx.x * 256;
  const float* Ab = attn + (size_t)bh * 2048 * 2048;
  const bf16* Vb  = V + (size_t)bh * 2048 * 64;
  v8f acc[2][4] = {};

  for (int k0 = 0; k0 < 2048; k0 += 32) {
    {  // A tile: 256x32, fp32 -> bf16 (one full row per thread)
      const float4* src =
          reinterpret_cast<const float4*>(Ab + (size_t)(row0 + tid) * 2048 + k0);
      bf16* dst = ldsA + tid * 32;
      cvt16_store(src, dst);
      cvt16_store(src + 4, dst + 16);
    }
    {  // B tile: V [k][d] -> N-major ldsBT[d][k] (transposing scatter)
      const int kk = tid >> 3, cb = (tid & 7) << 3;
      bf16 tmp[8];
      *reinterpret_cast<uint4*>(tmp) =
          *reinterpret_cast<const uint4*>(Vb + (size_t)(k0 + kk) * 64 + cb);
#pragma unroll
      for (int i = 0; i < 8; ++i) ldsBT[(cb + i) * 32 + kk] = tmp[i];
    }
    __syncthreads();
    v16bf a0 = frag_a(ldsA, wave * 32);
    v16bf a1 = frag_a(ldsA, wave * 32 + 16);
#pragma unroll
    for (int t = 0; t < 4; ++t) {
      v16bf bfr = frag_b(ldsBT, t * 16);
      acc[0][t] = wmma_bf16(a0, bfr, acc[0][t]);
      acc[1][t] = wmma_bf16(a1, bfr, acc[1][t]);
    }
    __syncthreads();
  }

  const int b = bh / 12, h = bh % 12;
  const int nl = lane & 15, mh = (lane >> 4) << 3;
#pragma unroll
  for (int u = 0; u < 2; ++u) {
#pragma unroll
    for (int t = 0; t < 4; ++t) {
      const int d = t * 16 + nl;
#pragma unroll
      for (int r = 0; r < 8; ++r) {
        const int q = row0 + wave * 32 + u * 16 + mh + r;
        AO[((size_t)(b * 2048 + q)) * 768 + h * 64 + d] = (bf16)acc[u][t][r];
      }
    }
  }
}

// ---------------------------------------------------------------------------
// Kernel 5: out = attn_out @ W_out^T + b_out (fp32 output)
//   M=4096, N=768, K=768. Block 128x128.
//   A tile (bf16 copy) goes through async DMA, overlapped with W conversion.
// ---------------------------------------------------------------------------
__global__ __launch_bounds__(256) void out_gemm(
    const bf16* __restrict__ AO, const float* __restrict__ W,
    const float* __restrict__ bias, float* __restrict__ out) {
  __shared__ alignas(32) bf16 ldsA[128 * 32];
  __shared__ alignas(32) bf16 ldsBT[128 * 32];
  const int tid = threadIdx.x, lane = tid & 31, wave = tid >> 5;
  const int wm = wave >> 1, wn = wave & 1;
  const int row0 = blockIdx.y * 128;
  const int col0 = blockIdx.x * 128;
  v8f acc[2][4] = {};

  for (int k0 = 0; k0 < 768; k0 += 32) {
    const int r = tid >> 1, c = (tid & 1) << 4;
    async_copy16bf(AO + (size_t)(row0 + r) * 768 + k0 + c, ldsA + r * 32 + c);
    cvt16_store(reinterpret_cast<const float4*>(W + (size_t)(col0 + r) * 768 + k0 + c),
                ldsBT + r * 32 + c);
    wait_async0();
    __syncthreads();
    v16bf a0 = frag_a(ldsA, wm * 32);
    v16bf a1 = frag_a(ldsA, wm * 32 + 16);
#pragma unroll
    for (int t = 0; t < 4; ++t) {
      v16bf bfr = frag_b(ldsBT, wn * 64 + t * 16);
      acc[0][t] = wmma_bf16(a0, bfr, acc[0][t]);
      acc[1][t] = wmma_bf16(a1, bfr, acc[1][t]);
    }
    __syncthreads();
  }

  const int nl = lane & 15, mh = (lane >> 4) << 3;
#pragma unroll
  for (int u = 0; u < 2; ++u) {
#pragma unroll
    for (int t = 0; t < 4; ++t) {
      const int f = col0 + wn * 64 + t * 16 + nl;
      const float bv = bias[f];
#pragma unroll
      for (int r = 0; r < 8; ++r) {
        const int tok = row0 + wm * 32 + u * 16 + mh + r;
        out[(size_t)tok * 768 + f] = acc[u][t][r] + bv;
      }
    }
  }
}

// ---------------------------------------------------------------------------
// Host launcher
// ---------------------------------------------------------------------------
extern "C" void kernel_launch(void* const* d_in, const int* in_sizes, int n_in,
                              void* d_out, int out_size, void* d_ws, size_t ws_size,
                              hipStream_t stream) {
  (void)in_sizes; (void)n_in; (void)out_size; (void)ws_size;
  const float* x            = (const float*)d_in[0];
  const unsigned char* mask = (const unsigned char*)d_in[1];
  const float* Wqkv         = (const float*)d_in[2];
  const float* bqkv         = (const float*)d_in[3];
  const float* Wout         = (const float*)d_in[4];
  const float* bout         = (const float*)d_in[5];

  float* out  = (float*)d_out;
  float* attn = out + (size_t)2 * 2048 * 768;  // [B,H,S,S] region of d_out

  const size_t QKVE = (size_t)2 * 12 * 2048 * 64;  // 3,145,728 elems
  bf16* qb = (bf16*)d_ws;
  bf16* kb = qb + QKVE;
  bf16* vb = kb + QKVE;
  bf16* ao = vb + QKVE;  // attn-out bf16 [B,S,E]

  qkv_gemm   <<<dim3(18, 32),     256, 0, stream>>>(x, Wqkv, bqkv, qb, kb, vb);
  scores_gemm<<<dim3(16, 16, 24), 256, 0, stream>>>(qb, kb, mask, attn);
  softmax_rows<<<dim3(49152),     256, 0, stream>>>(attn);
  av_gemm    <<<dim3(8, 24),      256, 0, stream>>>(attn, vb, ao);
  out_gemm   <<<dim3(6, 32),      256, 0, stream>>>(ao, Wout, bout, out);
}
